// LSTMDecoder_83013127897114
// MI455X (gfx1250) — compile-verified, matrix operand order checked by reading, not run on hardware
//
#include <hip/hip_runtime.h>
#include <hip/hip_bf16.h>
#include <math.h>

// Problem constants (match reference)
#define BB 64
#define SS 512
#define TT 256
#define VV 4096
#define DD 320
#define HH 320
#define AA 256

typedef __attribute__((ext_vector_type(16))) _Float16 v16h;
typedef __attribute__((ext_vector_type(8)))  _Float16 v8h;
typedef __attribute__((ext_vector_type(8)))  float    v8f;

static inline int cdiv(int a, int b) { return (a + b - 1) / b; }

// Load one 16x32 WMMA fragment row-slice for this lane (all-f16 data, two
// 16-byte vector loads). Fragment layout (ISA 7.12.2, 16-bit A 16x32):
// lanes 0-15 hold K {kk+0..7, kk+16..23}; lanes 16-31 hold K {kk+8..15, kk+24..31}.
__device__ __forceinline__ v16h load_frag(const _Float16* __restrict__ row,
                                          int klo, int khi)
{
    v8h lo = *(const v8h*)(row + klo);
    v8h hi = *(const v8h*)(row + khi);
    v16h r;
    #pragma unroll
    for (int j = 0; j < 8; ++j) { r[j] = lo[j]; r[j + 8] = hi[j]; }
    return r;
}

// ---------------------------------------------------------------------------
// WMMA GEMM:  C(MxN, ldC) = act( A(MxK f16) * W(NxK f16)^T + bias )
// Each wave owns a 32(M) x 64(N) strip of C: 2 A-fragments x 4 B-fragments
// -> 8 WMMAs per 32-wide K step (B fragment reused across both M tiles).
// Block = 128 threads = 4 waves = 32x256 strip per block.
// Requires M%32==0, N%64==0, K%32==0 (true for all call sites).
// Output: f32 to C if C!=null, else f16 to C16.
// ---------------------------------------------------------------------------
__global__ __launch_bounds__(128)
void gemm_wmma_kernel(const _Float16* __restrict__ A,
                      const _Float16* __restrict__ W,
                      const float* __restrict__ bias,
                      float* __restrict__ C,
                      _Float16* __restrict__ C16,
                      int M, int N, int K, int ldC, int act_tanh)
{
    const int lane = threadIdx.x & 31;
    const int wave = threadIdx.x >> 5;
    const int n0   = (blockIdx.x * 4 + wave) * 64;   // 64 output cols per wave
    const int m0   = blockIdx.y * 32;                // 32 output rows per wave
    if (n0 >= N) return;                             // wave-uniform exit

    const int half = lane >> 4;                      // 0: lanes 0-15, 1: lanes 16-31
    const int l16  = lane & 15;

    const _Float16* Arow0 = A + (size_t)(m0 +  0 + l16) * K;
    const _Float16* Arow1 = A + (size_t)(m0 + 16 + l16) * K;
    const _Float16* W0    = W + (size_t)(n0 +  0 + l16) * K;
    const _Float16* W1    = W + (size_t)(n0 + 16 + l16) * K;
    const _Float16* W2    = W + (size_t)(n0 + 32 + l16) * K;
    const _Float16* W3    = W + (size_t)(n0 + 48 + l16) * K;

    v8f a00 = {}, a01 = {}, a02 = {}, a03 = {};      // M tile 0, N tiles 0..3
    v8f a10 = {}, a11 = {}, a12 = {}, a13 = {};      // M tile 1, N tiles 0..3

    for (int kk = 0; kk < K; kk += 32) {
        const int klo = kk + half * 8;
        const int khi = kk + 16 + half * 8;

        if (kk + 32 < K) {                           // pull next K-chunk toward WGP$
            __builtin_prefetch(W0 + kk + 32, 0, 3);
            __builtin_prefetch(W1 + kk + 32, 0, 3);
            __builtin_prefetch(W2 + kk + 32, 0, 3);
            __builtin_prefetch(W3 + kk + 32, 0, 3);
        }

        v16h fa0 = load_frag(Arow0, klo, khi);
        v16h fa1 = load_frag(Arow1, klo, khi);
        v16h b0  = load_frag(W0, klo, khi);
        v16h b1  = load_frag(W1, klo, khi);
        v16h b2  = load_frag(W2, klo, khi);
        v16h b3  = load_frag(W3, klo, khi);

        a00 = __builtin_amdgcn_wmma_f32_16x16x32_f16(false, fa0, false, b0, (short)0, a00, false, false);
        a10 = __builtin_amdgcn_wmma_f32_16x16x32_f16(false, fa1, false, b0, (short)0, a10, false, false);
        a01 = __builtin_amdgcn_wmma_f32_16x16x32_f16(false, fa0, false, b1, (short)0, a01, false, false);
        a11 = __builtin_amdgcn_wmma_f32_16x16x32_f16(false, fa1, false, b1, (short)0, a11, false, false);
        a02 = __builtin_amdgcn_wmma_f32_16x16x32_f16(false, fa0, false, b2, (short)0, a02, false, false);
        a12 = __builtin_amdgcn_wmma_f32_16x16x32_f16(false, fa1, false, b2, (short)0, a12, false, false);
        a03 = __builtin_amdgcn_wmma_f32_16x16x32_f16(false, fa0, false, b3, (short)0, a03, false, false);
        a13 = __builtin_amdgcn_wmma_f32_16x16x32_f16(false, fa1, false, b3, (short)0, a13, false, false);
    }

    // C layout (ISA 7.12.2): VGPR r, lanes 0-15 -> row base+r, lanes 16-31 -> +8.
    v8f accs[2][4] = {{a00, a01, a02, a03}, {a10, a11, a12, a13}};
    #pragma unroll
    for (int mi = 0; mi < 2; ++mi) {
        #pragma unroll
        for (int i = 0; i < 4; ++i) {
            #pragma unroll
            for (int r = 0; r < 8; ++r) {
                int m = m0 + mi * 16 + r + half * 8;
                int n = n0 + i * 16 + l16;
                float v = accs[mi][i][r];
                if (bias) v += bias[n];
                if (act_tanh) v = tanhf(v);
                if (C)   C  [(size_t)m * ldC + n] = v;
                else     C16[(size_t)m * ldC + n] = (_Float16)v;
            }
        }
    }
}

// ---------------------------------------------------------------------------
// Small helper kernels
// ---------------------------------------------------------------------------
__global__ void cvt_f16_kernel(const float* __restrict__ src,
                               _Float16* __restrict__ dst, int n)
{
    int i = blockIdx.x * blockDim.x + threadIdx.x;
    if (i < n) dst[i] = (_Float16)src[i];
}

// Wg (1280 x 960) = [ W_ih (1280x640) | W_hh (1280x320) ] in f16
__global__ void pack_wg_kernel(const float* __restrict__ Wih,
                               const float* __restrict__ Whh,
                               _Float16* __restrict__ Wg)
{
    int i = blockIdx.x * blockDim.x + threadIdx.x;
    if (i >= 1280 * 960) return;
    int n = i / 960, k = i % 960;
    float v = (k < 640) ? Wih[n * 640 + k] : Whh[n * 320 + (k - 640)];
    Wg[i] = (_Float16)v;
}

__global__ void addvec_kernel(const float* a, const float* b, float* o, int n)
{
    int i = blockIdx.x * blockDim.x + threadIdx.x;
    if (i < n) o[i] = a[i] + b[i];
}

__global__ void zero_f32_kernel(float* p, int n)
{
    int i = blockIdx.x * blockDim.x + threadIdx.x;
    if (i < n) p[i] = 0.0f;
}

__global__ void zero_f16_kernel(_Float16* p, int n)
{
    int i = blockIdx.x * blockDim.x + threadIdx.x;
    if (i < n) p[i] = (_Float16)0.0f;
}

// m_mean[b,d] = mean_s memory[b,s,d]
__global__ void mean_kernel(const float* __restrict__ mem, float* __restrict__ out)
{
    int i = blockIdx.x * blockDim.x + threadIdx.x;
    if (i >= BB * DD) return;
    int b = i / DD, d = i % DD;
    const float* p = mem + (size_t)b * SS * DD + d;
    float s = 0.0f;
    for (int k = 0; k < SS; ++k) s += p[(size_t)k * DD];
    out[i] = s * (1.0f / SS);
}

// out[m,n] = tanh( sum_k A[m,k]*W[n,k] + bias[n] )   (one-time f32 init GEMV)
__global__ void init_state_kernel(const float* __restrict__ A,
                                  const float* __restrict__ W,
                                  const float* __restrict__ bias,
                                  float* __restrict__ out, int M, int N, int K)
{
    int i = blockIdx.x * blockDim.x + threadIdx.x;
    if (i >= M * N) return;
    int m = i / N, n = i % N;
    const float* a = A + (size_t)m * K;
    const float* w = W + (size_t)n * K;
    float s = bias[n];
    for (int k = 0; k < K; ++k) s += a[k] * w[k];
    out[i] = tanhf(s);
}

// lstm_full16[b, 0:960] = f16[ emb[tgt[b,t]], ctx_prev[b], h[b] ]
__global__ void assemble_in_kernel(const int* __restrict__ tgt,
                                   const float* __restrict__ emb,
                                   const _Float16* __restrict__ ctx16,
                                   const _Float16* __restrict__ h16,
                                   _Float16* __restrict__ out, int t)
{
    int i = blockIdx.x * blockDim.x + threadIdx.x;
    if (i >= BB * 960) return;
    int b = i / 960, k = i % 960;
    _Float16 v;
    if (k < DD) {
        int id = tgt[b * TT + t];
        v = (_Float16)emb[(size_t)id * DD + k];
    } else if (k < 2 * DD) {
        v = ctx16[b * DD + (k - DD)];
    } else {
        v = h16[b * HH + (k - 2 * DD)];
    }
    out[i] = v;
}

__device__ __forceinline__ float sigf(float x) { return 1.0f / (1.0f + __expf(-x)); }

// LSTM pointwise: gates (B x 4H f32, order i,f,g,o) -> c (f32, in place), h16 (f16)
__global__ void lstm_point_kernel(const float* __restrict__ gates,
                                  _Float16* __restrict__ h16,
                                  float* __restrict__ c)
{
    int i = blockIdx.x * blockDim.x + threadIdx.x;
    if (i >= BB * HH) return;
    int b = i / HH, j = i % HH;
    const float* g = gates + (size_t)b * 4 * HH;
    float gi = sigf(g[j]);
    float gf = sigf(g[HH + j]);
    float gg = tanhf(g[2 * HH + j]);
    float go = sigf(g[3 * HH + j]);
    float cn = gf * c[i] + gi * gg;
    float hn = go * tanhf(cn);
    c[i]   = cn;
    h16[i] = (_Float16)hn;
}

// score[b,s] = sum_a v[a] * tanh(hproj[b,a] + mproj[b,s,a] + cov[b,s]*wc[a])
// One wave per (b,s); A=256 -> 8 elems/lane; lane-shuffle reduction (wave32).
__global__ void attn_score_kernel(const float* __restrict__ hproj,
                                  const float* __restrict__ mproj,
                                  const float* __restrict__ cov,
                                  const float* __restrict__ wc,
                                  const float* __restrict__ vv,
                                  float* __restrict__ score)
{
    int lane = threadIdx.x & 31;
    int w = (blockIdx.x * blockDim.x + threadIdx.x) >> 5;   // (b*S + s)
    if (w >= BB * SS) return;
    int b = w / SS;
    float cv = cov[w];
    const float* mp = mproj + (size_t)w * AA;
    const float* hp = hproj + (size_t)b * AA;
    float sum = 0.0f;
    #pragma unroll
    for (int j = 0; j < AA / 32; ++j) {
        int a = j * 32 + lane;
        sum += vv[a] * tanhf(hp[a] + mp[a] + cv * wc[a]);
    }
    #pragma unroll
    for (int off = 16; off > 0; off >>= 1) sum += __shfl_xor(sum, off, 32);
    if (lane == 0) score[w] = sum;
}

// One block per batch b (256 threads): softmax over S=512, cov += alpha
__global__ __launch_bounds__(256)
void softmax_cov_kernel(const float* __restrict__ score,
                        float* __restrict__ alpha, float* __restrict__ cov)
{
    __shared__ float red_max[8];
    __shared__ float red_sum[8];
    int b = blockIdx.x, tid = threadIdx.x;
    int lane = tid & 31, wv = tid >> 5;
    const float* sc = score + (size_t)b * SS;
    float v0 = sc[tid], v1 = sc[tid + 256];

    float mx = fmaxf(v0, v1);
    #pragma unroll
    for (int o = 16; o > 0; o >>= 1) mx = fmaxf(mx, __shfl_xor(mx, o, 32));
    if (lane == 0) red_max[wv] = mx;
    __syncthreads();
    float m = red_max[0];
    #pragma unroll
    for (int i = 1; i < 8; ++i) m = fmaxf(m, red_max[i]);

    float e0 = __expf(v0 - m), e1 = __expf(v1 - m);
    float s = e0 + e1;
    #pragma unroll
    for (int o = 16; o > 0; o >>= 1) s += __shfl_xor(s, o, 32);
    if (lane == 0) red_sum[wv] = s;
    __syncthreads();
    float tot = 0.0f;
    #pragma unroll
    for (int i = 0; i < 8; ++i) tot += red_sum[i];
    float inv = 1.0f / tot;

    float a0 = e0 * inv, a1 = e1 * inv;
    alpha[(size_t)b * SS + tid]       = a0;
    alpha[(size_t)b * SS + tid + 256] = a1;
    cov[(size_t)b * SS + tid]        += a0;
    cov[(size_t)b * SS + tid + 256]  += a1;
}

// ctx16[b,d] = f16( sum_s alpha[b,s] * memory[b,s,d] )
__global__ void ctx_kernel(const float* __restrict__ alpha,
                           const float* __restrict__ mem,
                           _Float16* __restrict__ ctx16)
{
    int i = blockIdx.x * blockDim.x + threadIdx.x;
    if (i >= BB * DD) return;
    int b = i / DD, d = i % DD;
    const float* mp = mem + (size_t)b * SS * DD + d;
    const float* al = alpha + (size_t)b * SS;
    float s = 0.0f;
    for (int k = 0; k < SS; ++k) s += al[k] * mp[(size_t)k * DD];
    ctx16[i] = (_Float16)s;
}

// hcat16[b, 0:640] = [ h16[b], ctx16[b] ]
__global__ void hcat_kernel(const _Float16* __restrict__ h16,
                            const _Float16* __restrict__ ctx16,
                            _Float16* __restrict__ out)
{
    int i = blockIdx.x * blockDim.x + threadIdx.x;
    if (i >= BB * 2 * DD) return;
    int b = i / (2 * DD), k = i % (2 * DD);
    out[i] = (k < HH) ? h16[b * HH + k] : ctx16[b * DD + (k - HH)];
}

// ---------------------------------------------------------------------------
// Host launcher
// ---------------------------------------------------------------------------
extern "C" void kernel_launch(void* const* d_in, const int* in_sizes, int n_in,
                              void* d_out, int out_size, void* d_ws, size_t ws_size,
                              hipStream_t stream)
{
    const float* memory   = (const float*)d_in[0];
    const int*   tgt_ids  = (const int*)  d_in[1];
    const float* emb      = (const float*)d_in[2];
    const float* W_ih     = (const float*)d_in[3];
    const float* b_ih     = (const float*)d_in[4];
    const float* W_hh     = (const float*)d_in[5];
    const float* b_hh     = (const float*)d_in[6];
    const float* W_h      = (const float*)d_in[7];
    const float* W_m      = (const float*)d_in[8];
    const float* w_c      = (const float*)d_in[9];
    const float* v_vec    = (const float*)d_in[10];
    const float* W_ctx    = (const float*)d_in[11];
    const float* b_ctx    = (const float*)d_in[12];
    const float* W_out    = (const float*)d_in[13];
    const float* b_out    = (const float*)d_in[14];
    const float* W_init_h = (const float*)d_in[15];
    const float* b_init_h = (const float*)d_in[16];
    const float* W_init_c = (const float*)d_in[17];
    const float* b_init_c = (const float*)d_in[18];
    float* out = (float*)d_out;

    // Workspace carve-out (256B aligned slabs)
    unsigned char* ws = (unsigned char*)d_ws;
    size_t off = 0;
    auto alloc = [&](size_t bytes) -> void* {
        void* p = ws + off;
        off = (off + bytes + 255) & ~(size_t)255;
        return p;
    };
    _Float16* Wg16    = (_Float16*)alloc((size_t)1280 * 960 * 2);
    _Float16* Wh16    = (_Float16*)alloc((size_t)AA * HH * 2);
    _Float16* Wm16    = (_Float16*)alloc((size_t)AA * DD * 2);
    _Float16* Wctx16  = (_Float16*)alloc((size_t)DD * 2 * DD * 2);
    _Float16* Wout16  = (_Float16*)alloc((size_t)VV * DD * 2);
    _Float16* mem16   = (_Float16*)alloc((size_t)BB * SS * DD * 2);
    float*    bsum    = (float*)alloc(1280 * 4);
    float*    m_mean  = (float*)alloc((size_t)BB * DD * 4);
    float*    m_proj  = (float*)alloc((size_t)BB * SS * AA * 4);
    float*    htmp    = (float*)alloc((size_t)BB * HH * 4);
    float*    c_st    = (float*)alloc((size_t)BB * HH * 4);
    _Float16* h16     = (_Float16*)alloc((size_t)BB * HH * 2);
    _Float16* ctx16   = (_Float16*)alloc((size_t)BB * DD * 2);
    float*    hproj   = (float*)alloc((size_t)BB * AA * 4);
    float*    cov     = (float*)alloc((size_t)BB * SS * 4);
    float*    score   = (float*)alloc((size_t)BB * SS * 4);
    float*    alpha   = (float*)alloc((size_t)BB * SS * 4);
    _Float16* lfull16 = (_Float16*)alloc((size_t)BB * 960 * 2);
    float*    gates   = (float*)alloc((size_t)BB * 4 * HH * 4);
    _Float16* hcat16  = (_Float16*)alloc((size_t)BB * 2 * DD * 2);
    _Float16* dec16   = (_Float16*)alloc((size_t)BB * DD * 2);

    const int TB = 256;
    auto g1 = [&](int n) { return dim3(cdiv(n, TB)); };
    auto gemm_grid = [&](int M, int N) { return dim3(cdiv(N, 256), M / 32); };

    // ---- one-time preprocessing ----
    pack_wg_kernel<<<g1(1280 * 960), TB, 0, stream>>>(W_ih, W_hh, Wg16);
    cvt_f16_kernel<<<g1(AA * HH),      TB, 0, stream>>>(W_h,    Wh16,   AA * HH);
    cvt_f16_kernel<<<g1(AA * DD),      TB, 0, stream>>>(W_m,    Wm16,   AA * DD);
    cvt_f16_kernel<<<g1(DD * 2 * DD),  TB, 0, stream>>>(W_ctx,  Wctx16, DD * 2 * DD);
    cvt_f16_kernel<<<g1(VV * DD),      TB, 0, stream>>>(W_out,  Wout16, VV * DD);
    cvt_f16_kernel<<<g1(BB * SS * DD), TB, 0, stream>>>(memory, mem16,  BB * SS * DD);
    addvec_kernel<<<g1(1280), TB, 0, stream>>>(b_ih, b_hh, bsum, 1280);

    mean_kernel<<<g1(BB * DD), TB, 0, stream>>>(memory, m_mean);
    init_state_kernel<<<g1(BB * HH), TB, 0, stream>>>(m_mean, W_init_h, b_init_h, htmp, BB, HH, DD);
    cvt_f16_kernel<<<g1(BB * HH), TB, 0, stream>>>(htmp, h16, BB * HH);
    init_state_kernel<<<g1(BB * HH), TB, 0, stream>>>(m_mean, W_init_c, b_init_c, c_st, BB, HH, DD);
    zero_f16_kernel<<<g1(BB * DD), TB, 0, stream>>>(ctx16, BB * DD);
    zero_f32_kernel<<<g1(BB * SS), TB, 0, stream>>>(cov, BB * SS);

    // m_proj = memory @ W_m^T : (B*S, A) = (32768, 256), K = 320
    gemm_wmma_kernel<<<gemm_grid(BB * SS, AA), 128, 0, stream>>>(
        mem16, Wm16, nullptr, m_proj, nullptr, BB * SS, AA, DD, AA, 0);

    // ---- decode loop ----
    for (int t = 0; t < TT; ++t) {
        assemble_in_kernel<<<g1(BB * 960), TB, 0, stream>>>(tgt_ids, emb, ctx16, h16, lfull16, t);

        // gates = lfull @ [W_ih|W_hh]^T + (b_ih+b_hh) : (64, 1280), K = 960
        gemm_wmma_kernel<<<gemm_grid(BB, 1280), 128, 0, stream>>>(
            lfull16, Wg16, bsum, gates, nullptr, BB, 1280, 960, 1280, 0);

        lstm_point_kernel<<<g1(BB * HH), TB, 0, stream>>>(gates, h16, c_st);

        // hproj = h @ W_h^T : (64, 256), K = 320
        gemm_wmma_kernel<<<gemm_grid(BB, AA), 128, 0, stream>>>(
            h16, Wh16, nullptr, hproj, nullptr, BB, AA, HH, AA, 0);

        // coverage attention score: one wave per (b,s)
        attn_score_kernel<<<dim3((BB * SS) / 8), TB, 0, stream>>>(
            hproj, m_proj, cov, w_c, v_vec, score);

        softmax_cov_kernel<<<dim3(BB), 256, 0, stream>>>(score, alpha, cov);

        ctx_kernel<<<g1(BB * DD), TB, 0, stream>>>(alpha, memory, ctx16);

        hcat_kernel<<<g1(BB * 2 * DD), TB, 0, stream>>>(h16, ctx16, hcat16);

        // dec = tanh(hcat @ W_ctx^T + b_ctx) : (64, 320), K = 640 -> f16
        gemm_wmma_kernel<<<gemm_grid(BB, DD), 128, 0, stream>>>(
            hcat16, Wctx16, b_ctx, nullptr, dec16, BB, DD, 2 * DD, DD, 1);

        // logits = dec @ W_out^T + b_out -> out[:, t, :]  (ldC = T*V)
        gemm_wmma_kernel<<<gemm_grid(BB, VV), 128, 0, stream>>>(
            dec16, Wout16, b_out, out + (size_t)t * VV, nullptr, BB, VV, DD, TT * VV, 0);
    }
}
